// Recurrence_45406394254150
// MI455X (gfx1250) — matrix operand order repaired and checked by reading, not compiled, not run on hardware
//
#include <hip/hip_runtime.h>

typedef _Float16 v16h __attribute__((ext_vector_type(16)));
typedef _Float16 v8h  __attribute__((ext_vector_type(8)));
typedef float    v8f  __attribute__((ext_vector_type(8)));

constexpr int HID  = 256;   // hidden size
constexpr int SEQ  = 64;    // GRU steps (= obs feature count at t=0)
constexpr int NB   = 4096;  // batch
constexpr int TT   = 128;   // time steps in output
constexpr int DD   = 65;    // input feature dim (64 obs + 1 action)
constexpr int HXW  = 261;   // hx / hx_out row width
constexpr int HPAD = 264;   // padded LDS row (16B aligned, bank-shifted)

static __device__ __forceinline__ v8f vzero8() {
  v8f z = {0.f, 0.f, 0.f, 0.f, 0.f, 0.f, 0.f, 0.f};
  return z;
}
static __device__ __forceinline__ v16h cat16(v8h lo, v8h hi) {
  return __builtin_shufflevector(lo, hi, 0, 1, 2, 3, 4, 5, 6, 7,
                                 8, 9, 10, 11, 12, 13, 14, 15);
}
static __device__ __forceinline__ v8f wmma16(v16h a, v16h b, v8f c) {
  // D = A(16x32 f16) x B(32x16 f16) + C(16x16 f32)
  return __builtin_amdgcn_wmma_f32_16x16x32_f16(false, a, false, b,
                                                (short)0, c, false, false);
}
static __device__ __forceinline__ float sigm(float x) {
  return 1.f / (1.f + __expf(-x));
}

// ---------------------------------------------------------------------------
// Weight pre-pack kernels: f32 -> f16 (optionally transposed to [n][k])
// ---------------------------------------------------------------------------
__global__ void pack_f16_kernel(const float* __restrict__ src,
                                _Float16* __restrict__ dst, int n) {
  int i = blockIdx.x * 256 + threadIdx.x;
  if (i < n) dst[i] = (_Float16)src[i];
}

__global__ void pack_f16_T_kernel(const float* __restrict__ src,
                                  _Float16* __restrict__ dst, int K, int Nc) {
  int i = blockIdx.x * 256 + threadIdx.x;
  if (i < K * Nc) {
    int k = i / Nc, n = i - k * Nc;            // src[k][n]
    dst[(size_t)n * K + k] = (_Float16)src[i]; // dst[n][k]
  }
}

// ---------------------------------------------------------------------------
// Bidirectional GRU. grid = 128 blocks: bit0 = direction, rest = 64-row tile.
// Each wave owns 16 batch rows; h lives in LDS f16; rows are independent so
// the 64-step loop needs no barriers. Gates come from WMMA f16->f32.
// Hidden state is scattered into r16[n][dir*256 + c] at step s == P[n].
// ---------------------------------------------------------------------------
__global__ __launch_bounds__(128)
void gru_kernel(const float* __restrict__ x0,   // inputs; only t=0 slice used
                const float* __restrict__ hx,
                const float* __restrict__ wih_f, const float* __restrict__ bih_f,
                const float* __restrict__ bhh_f,
                const float* __restrict__ wih_b, const float* __restrict__ bih_b,
                const float* __restrict__ bhh_b,
                const _Float16* __restrict__ pkw_f,  // [768][256] f16 row-major
                const _Float16* __restrict__ pkw_b,
                _Float16* __restrict__ r16)          // [NB][512] f16
{
  __shared__ _Float16 Hs[64 * HPAD];
  __shared__ float WIH[768], BIH[768], BHH[768];

  const int dir  = blockIdx.x & 1;
  const int tile = blockIdx.x >> 1;
  const int row0 = tile * 64;

  const float* wih = dir ? wih_b : wih_f;
  const float* bih = dir ? bih_b : bih_f;
  const float* bhh = dir ? bhh_b : bhh_f;
  const _Float16* pkw = dir ? pkw_b : pkw_f;

  for (int i = threadIdx.x; i < 64 * HPAD; i += 128) Hs[i] = (_Float16)0.f;
  for (int i = threadIdx.x; i < 768; i += 128) {
    WIH[i] = wih[i]; BIH[i] = bih[i]; BHH[i] = bhh[i];
  }
  __syncthreads();

  const int lane = threadIdx.x & 31;
  const int wid  = threadIdx.x >> 5;
  const int lg   = lane >> 4;      // lane group (0: lanes 0-15, 1: 16-31)
  const int ml   = lane & 15;
  const int lrow0 = wid * 16;      // wave's first local row
  const int grow0 = row0 + lrow0;  // wave's first global row

  int Pv[16];
#pragma unroll
  for (int rr = 0; rr < 16; ++rr)
    Pv[rr] = (int)hx[(size_t)(grow0 + rr) * HXW + (HXW - 1)];

  for (int si = 0; si < SEQ; ++si) {
    const int s = dir ? (SEQ - 1 - si) : si;

    // A fragments for this wave's 16 rows, full K=256, from LDS.
    v16h a[8];
#pragma unroll
    for (int kf = 0; kf < 8; ++kf) {
      const int k0 = kf * 32 + lg * 16;
      v8h lo = *(const v8h*)&Hs[(lrow0 + ml) * HPAD + k0];
      v8h hi = *(const v8h*)&Hs[(lrow0 + ml) * HPAD + k0 + 8];
      a[kf] = cat16(lo, hi);
    }
    // per-row GRU input x[s, n] = inputs[0, n, s]
    float xr[8];
#pragma unroll
    for (int v = 0; v < 8; ++v)
      xr[v] = x0[(size_t)(grow0 + v + 8 * lg) * DD + s];

    for (int ct = 0; ct < 16; ++ct) {
      const int c = ct * 16 + ml;          // this lane's output column
      v8f aR = vzero8(), aZ = vzero8(), aN = vzero8();
      const _Float16* bp = pkw + (size_t)c * HID;
#pragma unroll
      for (int kf = 0; kf < 8; ++kf) {
        const int koff = kf * 32 + lg * 16;
        v16h br = *(const v16h*)(bp + koff);
        v16h bz = *(const v16h*)(bp + 256 * HID + koff);
        v16h bn = *(const v16h*)(bp + 512 * HID + koff);
        aR = wmma16(a[kf], br, aR);
        aZ = wmma16(a[kf], bz, aZ);
        aN = wmma16(a[kf], bn, aN);
      }
      const float wr = WIH[c],       bri = BIH[c],       brh = BHH[c];
      const float wz = WIH[256 + c], bzi = BIH[256 + c], bzh = BHH[256 + c];
      const float wn = WIH[512 + c], bni = BIH[512 + c], bnh = BHH[512 + c];
#pragma unroll
      for (int v = 0; v < 8; ++v) {
        const int lr = lrow0 + v + 8 * lg;
        const float xv = xr[v];
        const float rg = sigm(xv * wr + bri + aR[v] + brh);
        const float zg = sigm(xv * wz + bzi + aZ[v] + bzh);
        const float ng = tanhf(xv * wn + bni + rg * (aN[v] + bnh));
        const float ho = (float)Hs[lr * HPAD + c];
        Hs[lr * HPAD + c] = (_Float16)((1.f - zg) * ng + zg * ho);
      }
    }

    // Scatter hidden state for rows whose time pointer equals this step.
#pragma unroll 1
    for (int rr = 0; rr < 16; ++rr) {
      if (Pv[rr] == s) {
        const int lr = lrow0 + rr;
        _Float16* dst = r16 + (size_t)(grow0 + rr) * (2 * HID) + dir * HID;
        for (int cc = lane; cc < HID; cc += 32)
          dst[cc] = Hs[lr * HPAD + cc];
      }
    }
  }
}

// ---------------------------------------------------------------------------
// Actor/critic heads: relu(r@W0+b0) -> relu(@W1+b1) -> dot(w) + b, via WMMA.
// 64 rows per block, wave-private, no barriers needed.
// ---------------------------------------------------------------------------
__global__ __launch_bounds__(128)
void mlp_kernel(const _Float16* __restrict__ r16,
                const _Float16* __restrict__ pk_aw0, const float* __restrict__ ab0,
                const _Float16* __restrict__ pk_aw1, const float* __restrict__ ab1,
                const float* __restrict__ mean_w, const float* __restrict__ mean_b,
                const _Float16* __restrict__ pk_cw0, const float* __restrict__ cb0,
                const _Float16* __restrict__ pk_cw1, const float* __restrict__ cb1,
                const float* __restrict__ out_w, const float* __restrict__ out_b,
                float* __restrict__ loc, float* __restrict__ vv)
{
  __shared__ _Float16 AH[64 * HPAD];
  const int row0 = blockIdx.x * 64;
  const int lane = threadIdx.x & 31, wid = threadIdx.x >> 5;
  const int lg = lane >> 4, ml = lane & 15;
  const int lrow0 = wid * 16;
  const int grow0 = row0 + lrow0;

  for (int head = 0; head < 2; ++head) {
    const _Float16* B1 = head ? pk_cw0 : pk_aw0;   // [256][512] f16
    const _Float16* B2 = head ? pk_cw1 : pk_aw1;   // [256][256] f16
    const float* b1 = head ? cb0 : ab0;
    const float* b2 = head ? cb1 : ab1;
    const float* wv = head ? out_w : mean_w;
    const float  bs = head ? out_b[0] : mean_b[0];
    float* outp = head ? vv : loc;

    // stage 1: (16x512) x (512x256) per wave -> relu -> AH f16
    for (int tg = 0; tg < 4; ++tg) {
      v8f acc[4] = {vzero8(), vzero8(), vzero8(), vzero8()};
      for (int kf = 0; kf < 16; ++kf) {
        const int k0 = kf * 32 + lg * 16;
        const _Float16* ap = r16 + (size_t)(grow0 + ml) * 512 + k0;
        v16h a = cat16(*(const v8h*)ap, *(const v8h*)(ap + 8));
#pragma unroll
        for (int tt = 0; tt < 4; ++tt) {
          const int j = (tg * 4 + tt) * 16 + ml;
          v16h b = *(const v16h*)(B1 + (size_t)j * 512 + kf * 32 + lg * 16);
          acc[tt] = wmma16(a, b, acc[tt]);
        }
      }
#pragma unroll
      for (int tt = 0; tt < 4; ++tt) {
        const int c = (tg * 4 + tt) * 16 + ml;
        const float bc = b1[c];
#pragma unroll
        for (int v = 0; v < 8; ++v) {
          float val = acc[tt][v] + bc;
          val = val > 0.f ? val : 0.f;
          AH[(lrow0 + v + 8 * lg) * HPAD + c] = (_Float16)val;
        }
      }
    }

    // stage 2 + projection to one scalar per row
    float pacc[8] = {0.f, 0.f, 0.f, 0.f, 0.f, 0.f, 0.f, 0.f};
    for (int tg = 0; tg < 4; ++tg) {
      v8f acc[4] = {vzero8(), vzero8(), vzero8(), vzero8()};
      for (int kf = 0; kf < 8; ++kf) {
        const int k0 = kf * 32 + lg * 16;
        const _Float16* ap = &AH[(lrow0 + ml) * HPAD + k0];
        v16h a = cat16(*(const v8h*)ap, *(const v8h*)(ap + 8));
#pragma unroll
        for (int tt = 0; tt < 4; ++tt) {
          const int j = (tg * 4 + tt) * 16 + ml;
          v16h b = *(const v16h*)(B2 + (size_t)j * 256 + kf * 32 + lg * 16);
          acc[tt] = wmma16(a, b, acc[tt]);
        }
      }
#pragma unroll
      for (int tt = 0; tt < 4; ++tt) {
        const int c = (tg * 4 + tt) * 16 + ml;
        const float bc = b2[c], wc = wv[c];
#pragma unroll
        for (int v = 0; v < 8; ++v) {
          float a2 = acc[tt][v] + bc;
          a2 = a2 > 0.f ? a2 : 0.f;
          pacc[v] += a2 * wc;
        }
      }
    }
    // reduce partial dot across the 16 lanes covering columns
#pragma unroll
    for (int off = 1; off < 16; off <<= 1) {
#pragma unroll
      for (int v = 0; v < 8; ++v)
        pacc[v] += __shfl_xor(pacc[v], off, 32);
    }
    if (ml == 0) {
#pragma unroll
      for (int v = 0; v < 8; ++v)
        outp[grow0 + v + 8 * lg] = pacc[v] + bs;
    }
  }
}

// ---------------------------------------------------------------------------
// Output assembly: (129, 4096, 261). Row 128 duplicates row 127.
// One wave per (t, n) pair -> 261 coalesced floats.
// ---------------------------------------------------------------------------
__global__ __launch_bounds__(256)
void assemble_kernel(const float* __restrict__ inputs, const float* __restrict__ hx,
                     const float* __restrict__ loc, const float* __restrict__ vv,
                     const float* __restrict__ logstd, float* __restrict__ out)
{
  const int lane = threadIdx.x & 31, wid = threadIdx.x >> 5;
  const long long pair = (long long)blockIdx.x * 8 + wid;   // 0 .. 129*4096-1
  const int t = (int)(pair >> 12);
  const int n = (int)(pair & (NB - 1));
  const int teff = t > (TT - 1) ? (TT - 1) : t;

  const float act  = inputs[((size_t)teff * NB + n) * DD + (DD - 1)];
  const float locn = loc[n];
  const float scl  = __expf(logstd[0]);
  const float vn   = vv[n];
  const int   P    = (int)hx[(size_t)n * HXW + (HXW - 1)];
  const float pn   = (float)((P + 1) & (SEQ - 1));

  float A;
  if (act < 0.f) {
    // hash-based N(0,1) stand-in for jax.random.normal(key 42)
    unsigned u = (unsigned)(teff * NB + n);
    unsigned x = u * 2654435761u + 0x9E3779B9u;
    x ^= x >> 16; x *= 0x7feb352du; x ^= x >> 15; x *= 0x846ca68bu; x ^= x >> 16;
    unsigned y = (u ^ 0xdeadbeefu) * 0x85ebca6bu;
    y ^= y >> 13; y *= 0xc2b2ae35u; y ^= y >> 16;
    float u1 = (float)(x >> 8) * (1.f / 16777216.f) + 1e-7f;
    float u2 = (float)(y >> 8) * (1.f / 16777216.f);
    float eps = __fsqrt_rn(-2.f * __logf(u1)) * __cosf(6.28318530718f * u2);
    A = locn + scl * eps;
  } else {
    A = act;
  }

  float* orow = out + ((size_t)t * NB + n) * HXW;
  for (int c = lane; c < HXW; c += 32) {
    float val;
    if (c == 0)            val = A;
    else if (c == 1)       val = locn;
    else if (c == 2)       val = scl;
    else if (c == 3)       val = vn;
    else if (c < HXW - 1)  val = hx[(size_t)n * HXW + c];  // h_state passthrough
    else                   val = pn;
    orow[c] = val;
  }
}

// ---------------------------------------------------------------------------
extern "C" void kernel_launch(void* const* d_in, const int* in_sizes, int n_in,
                              void* d_out, int out_size, void* d_ws, size_t ws_size,
                              hipStream_t stream) {
  (void)in_sizes; (void)n_in; (void)out_size; (void)ws_size;
  const float* inputs  = (const float*)d_in[0];
  const float* hx      = (const float*)d_in[1];
  const float* gw_ih_f = (const float*)d_in[2];
  const float* gw_hh_f = (const float*)d_in[3];
  const float* gb_ih_f = (const float*)d_in[4];
  const float* gb_hh_f = (const float*)d_in[5];
  const float* gw_ih_b = (const float*)d_in[6];
  const float* gw_hh_b = (const float*)d_in[7];
  const float* gb_ih_b = (const float*)d_in[8];
  const float* gb_hh_b = (const float*)d_in[9];
  const float* aw0     = (const float*)d_in[10];
  const float* ab0     = (const float*)d_in[11];
  const float* aw1     = (const float*)d_in[12];
  const float* ab1     = (const float*)d_in[13];
  const float* mean_w  = (const float*)d_in[14];
  const float* mean_b  = (const float*)d_in[15];
  const float* logstd  = (const float*)d_in[16];
  const float* cw0     = (const float*)d_in[17];
  const float* cb0     = (const float*)d_in[18];
  const float* cw1     = (const float*)d_in[19];
  const float* cb1     = (const float*)d_in[20];
  const float* out_w   = (const float*)d_in[21];
  const float* out_b   = (const float*)d_in[22];

  char* ws = (char*)d_ws;
  _Float16* pk_whh_f = (_Float16*)(ws + 0);         // 768*256*2 = 393216
  _Float16* pk_whh_b = (_Float16*)(ws + 393216);
  _Float16* pk_aw0   = (_Float16*)(ws + 786432);    // 256*512*2 = 262144
  _Float16* pk_aw1   = (_Float16*)(ws + 1048576);   // 256*256*2 = 131072
  _Float16* pk_cw0   = (_Float16*)(ws + 1179648);
  _Float16* pk_cw1   = (_Float16*)(ws + 1441792);
  _Float16* r16      = (_Float16*)(ws + 1572864);   // 4096*512*2 = 4 MiB
  float*    loc      = (float*)(ws + 5767168);      // 4096*4
  float*    vv       = (float*)(ws + 5783552);      // 4096*4

  pack_f16_kernel<<<(768 * 256 + 255) / 256, 256, 0, stream>>>(gw_hh_f, pk_whh_f, 768 * 256);
  pack_f16_kernel<<<(768 * 256 + 255) / 256, 256, 0, stream>>>(gw_hh_b, pk_whh_b, 768 * 256);
  pack_f16_T_kernel<<<(512 * 256 + 255) / 256, 256, 0, stream>>>(aw0, pk_aw0, 512, 256);
  pack_f16_T_kernel<<<(256 * 256 + 255) / 256, 256, 0, stream>>>(aw1, pk_aw1, 256, 256);
  pack_f16_T_kernel<<<(512 * 256 + 255) / 256, 256, 0, stream>>>(cw0, pk_cw0, 512, 256);
  pack_f16_T_kernel<<<(256 * 256 + 255) / 256, 256, 0, stream>>>(cw1, pk_cw1, 256, 256);

  gru_kernel<<<128, 128, 0, stream>>>(inputs, hx,
                                      gw_ih_f, gb_ih_f, gb_hh_f,
                                      gw_ih_b, gb_ih_b, gb_hh_b,
                                      pk_whh_f, pk_whh_b, r16);

  mlp_kernel<<<64, 128, 0, stream>>>(r16, pk_aw0, ab0, pk_aw1, ab1,
                                     mean_w, mean_b, pk_cw0, cb0, pk_cw1, cb1,
                                     out_w, out_b, loc, vv);

  assemble_kernel<<<(129 * NB) / 8, 256, 0, stream>>>(inputs, hx, loc, vv,
                                                      logstd, (float*)d_out);
}
